// Sequence_61400852464413
// MI455X (gfx1250) — compile-verified, matrix operand order checked by reading, not run on hardware
//
#include <hip/hip_runtime.h>
#include <hip/hip_bf16.h>
#include <stddef.h>

// ---------------------------------------------------------------------------
// LSTM stack for MI455X (gfx1250): persistent per-batch-tile workgroups,
// bf16 WMMA (v_wmma_f32_16x16x32_bf16) for all recurrent/input GEMMs,
// gates + cell state kept in registers in the WMMA C/D layout.
//
// NINP=64, NH=512, NOUT=1, T=64, B=2048.  128 blocks x 512 threads (16 waves).
// Each block owns 16 batch rows; no inter-block synchronization needed.
//
// Round-2: laundered the weight pointer per time step to stop LICM from
//          hoisting weight loads and spilling them to scratch. Worked, but
//          killed address-space inference -> flat_load_b128 + combined
//          loadcnt/dscnt waits.
// Round-3: launder a zero *offset* (SGPR) instead of the pointer, keeping
//          the access provably global -> global_load_b128 with SGPR base,
//          LOADcnt-only waits, while still preventing cross-step hoisting.
// ---------------------------------------------------------------------------

typedef __bf16 bf16_t;
typedef bf16_t v16bf __attribute__((ext_vector_type(16)));
typedef bf16_t v8bf  __attribute__((ext_vector_type(8)));
typedef float  v8f   __attribute__((ext_vector_type(8)));

#define NINP   64
#define NH     512
#define T_STEPS 64
#define ROW_STRIDE 4097      // T*NINP + 1
#define XPAD   96            // padded input width (covers 64 and 65)
#define NTILES 128           // 4*NH / 16 output column tiles

// Opaque zero in an SGPR: added to a pointer it defeats LICM across the time
// loop (loads stay in-loop, streamed from L2) without breaking global
// address-space inference (unlike laundering the pointer itself).
__device__ __forceinline__ size_t opaque_zero() {
    size_t x = 0;
    asm volatile("" : "+s"(x));
    return x;
}

// ---------------------------------------------------------------------------
// Weight packing: store W (4NH x K, row-major, g[n] = sum_k in[k]*W[n][k])
// as bf16 B-fragments for v_wmma_f32_16x16x32_bf16.
// B is K(32) x N(16); per ISA 7.12.2 (mirror of 16-bit A layout):
//   lane l (<16):  col = l,     K = {0..7}  in VGPR0-3, {16..23} in VGPR4-7
//   lane l (>=16): col = l-16,  K = {8..15} in VGPR0-3, {24..31} in VGPR4-7
// Packed layout: dst[(nt*KT + kb)*32 + lane][16 bf16], contiguous per lane.
// K-space = [input part: kin_blocks*32 (zero padded past kin_actual)] ++
//           [recurrent part: Whh, K=512].
// ---------------------------------------------------------------------------
__global__ void pack_weights_kernel(const float* __restrict__ Wih,
                                    const float* __restrict__ Whh,
                                    int kin_actual, int kin_blocks, int kt,
                                    bf16_t* __restrict__ dst)
{
    size_t total = (size_t)NTILES * kt * 512;      // 32 lanes * 16 elems per block
    size_t id = (size_t)blockIdx.x * blockDim.x + threadIdx.x;
    if (id >= total) return;

    int e    = (int)(id & 15);
    int lane = (int)((id >> 4) & 31);
    int kb   = (int)((id >> 9) % (size_t)kt);
    int nt   = (int)(id / ((size_t)kt << 9));

    int v = e >> 1, p = e & 1;
    int kl = (v < 4 ? v * 2 + p : 16 + (v - 4) * 2 + p) + ((lane >= 16) ? 8 : 0);
    int col = nt * 16 + (lane & 15);
    int k   = kb * 32 + kl;

    float val;
    if (kb < kin_blocks) {
        val = (k < kin_actual) ? Wih[(size_t)col * kin_actual + k] : 0.0f;
    } else {
        int kk = k - kin_blocks * 32;
        val = Whh[(size_t)col * NH + kk];
    }
    dst[id] = (bf16_t)val;
}

__global__ void pack_bias_kernel(const float* __restrict__ bihP, const float* __restrict__ bhhP,
                                 const float* __restrict__ bih1, const float* __restrict__ bhh1,
                                 const float* __restrict__ bih2, const float* __restrict__ bhh2,
                                 float* __restrict__ bsum)
{
    int i = blockIdx.x * blockDim.x + threadIdx.x;
    if (i < 2048)       bsum[i] = bihP[i] + bhhP[i];
    else if (i < 4096)  bsum[i] = bih1[i - 2048] + bhh1[i - 2048];
    else if (i < 6144)  bsum[i] = bih2[i - 4096] + bhh2[i - 4096];
}

// ---------------------------------------------------------------------------
// device helpers
// ---------------------------------------------------------------------------
__device__ __forceinline__ float sigm_f(float x) {
    return 1.0f / (1.0f + __expf(-x));
}
__device__ __forceinline__ float tanh_f(float x) {
    float e = __expf(2.0f * x);
    return (e - 1.0f) / (e + 1.0f);
}

// A-fragment (16x32 bf16) from LDS, row-major with row stride `astride`:
//   lane l: row = l&15; K-group = l>>4; chunks at k0 and k0+16 (8 bf16 each).
__device__ __forceinline__ v16bf load_afrag(const bf16_t* base, int astride,
                                            int lane, int kb)
{
    int row = lane & 15;
    int kg  = lane >> 4;
    int k0  = kb * 32 + kg * 8;
    union { v16bf v; v8bf h[2]; } u;
    u.h[0] = *(const v8bf*)(base + row * astride + k0);
    u.h[1] = *(const v8bf*)(base + row * astride + k0 + 16);
    return u.v;
}

// Accumulate nkb K-blocks of A(LDS) x B(packed global) into 8 column-tile accs.
__device__ __forceinline__ void mm_acc(const bf16_t* __restrict__ Wp, int KT,
                                       int kb0, int nkb,
                                       const bf16_t* Ab, int astride,
                                       int lane, const int* tiles, v8f* acc)
{
    for (int kk = 0; kk < nkb; ++kk) {
        int kb = kb0 + kk;
        v16bf a = load_afrag(Ab, astride, lane, kk);
        if (kk + 1 < nkb) {   // prefetch next K-block of first tile into caches
            __builtin_prefetch(Wp + ((size_t)(tiles[0] * KT + kb + 1) * 32 + lane) * 16, 0, 3);
        }
#pragma unroll
        for (int j = 0; j < 8; ++j) {
            const bf16_t* bp = Wp + ((size_t)(tiles[j] * KT + kb) * 32 + lane) * 16;
            v16bf b = *(const v16bf*)bp;
            acc[j] = __builtin_amdgcn_wmma_f32_16x16x32_bf16(
                         false, a, false, b, (short)0, acc[j], false, false);
        }
    }
}

// Gate math in the WMMA C/D register layout. acc[j], j = gate*2 + sub,
// gate in {i,f,g,o}; sub picks hidden unit n or n+16 for this lane.
__device__ __forceinline__ void lstm_gates(const v8f* acc, const float* bias,
                                           float c[2][8], float hout[2][8])
{
#pragma unroll
    for (int s = 0; s < 2; ++s) {
#pragma unroll
        for (int r = 0; r < 8; ++r) {
            float gi = acc[0 + s][r] + bias[0 + s];
            float gf = acc[2 + s][r] + bias[2 + s];
            float gg = acc[4 + s][r] + bias[4 + s];
            float go = acc[6 + s][r] + bias[6 + s];
            float iv = sigm_f(gi);
            float fv = sigm_f(gf);
            float gv = tanh_f(gg);
            float ov = sigm_f(go);
            float cn = fv * c[s][r] + iv * gv;
            c[s][r] = cn;
            hout[s][r] = ov * tanh_f(cn);
        }
    }
}

// Scatter h (register tile layout) back to LDS row-major bf16 [16][512].
__device__ __forceinline__ void store_h(bf16_t* hl, int lane, int w,
                                        const float hout[2][8])
{
    int col0  = w * 32 + (lane & 15);
    int rbase = (lane >> 4) * 8;
#pragma unroll
    for (int r = 0; r < 8; ++r) {
        hl[(rbase + r) * NH + col0]      = (bf16_t)hout[0][r];
        hl[(rbase + r) * NH + col0 + 16] = (bf16_t)hout[1][r];
    }
}

// Partial h . Wl per batch row via LDS float atomics.
__device__ __forceinline__ void dot_partial(float* sdot, int lane,
                                            float wl0, float wl1,
                                            const float hout[2][8])
{
    int rbase = (lane >> 4) * 8;
#pragma unroll
    for (int r = 0; r < 8; ++r) {
        atomicAdd(&sdot[rbase + r], hout[0][r] * wl0 + hout[1][r] * wl1);
    }
}

// ---------------------------------------------------------------------------
// Main persistent kernel: one block = 16 batch rows, all 64+64 time steps.
// ---------------------------------------------------------------------------
__global__ __launch_bounds__(512, 1)
void lstm_stack_kernel(const float* __restrict__ input,
                       const bf16_t* __restrict__ WpP,   // pia layer, KT=18
                       const bf16_t* __restrict__ Wp1,   // layer1,    KT=19
                       const bf16_t* __restrict__ Wp2,   // layer2,    KT=32
                       const float*  __restrict__ bsum,  // [3][2048]
                       const float*  __restrict__ Wl,    // [512]
                       const float*  __restrict__ blp,   // [1]
                       float* __restrict__ out)          // [B][65]
{
    __shared__ __align__(16) bf16_t s_x [16 * XPAD];   // x_t (+pia) tile, bf16
    __shared__ __align__(16) bf16_t s_hA[16 * NH];     // h_pia / h1
    __shared__ __align__(16) bf16_t s_hB[16 * NH];     // h2
    __shared__ float s_pia[T_STEPS * 16];              // pia scalars per (t,row)
    __shared__ float s_dot[16];                        // output-head reduction

    const int tid  = threadIdx.x;
    const int lane = tid & 31;
    const int w    = tid >> 5;                 // wave 0..15
    const int rbase = blockIdx.x * 16;         // batch rows [rbase, rbase+16)

    // Each wave owns matching 32-unit slices of all four gates:
    // tile(g, s) = g*32 + w*2 + s  -> hidden units n in [w*32, w*32+32).
    int tiles[8];
#pragma unroll
    for (int g = 0; g < 4; ++g) {
        tiles[g * 2]     = g * 32 + w * 2;
        tiles[g * 2 + 1] = g * 32 + w * 2 + 1;
    }
    const float wl0 = Wl[w * 32 + (lane & 15)];
    const float wl1 = Wl[w * 32 + 16 + (lane & 15)];
    const float bl0 = blp[0];

    float biasP[8], bias1[8], bias2[8];
#pragma unroll
    for (int j = 0; j < 8; ++j) {
        int n = tiles[j] * 16 + (lane & 15);
        biasP[j] = bsum[n];
        bias1[j] = bsum[2048 + n];
        bias2[j] = bsum[4096 + n];
    }

    for (int i = tid; i < 16 * NH; i += 512) {
        s_hA[i] = (bf16_t)0.0f;
        s_hB[i] = (bf16_t)0.0f;
    }
    if (tid < 16) s_dot[tid] = 0.0f;
    float scale = 0.0f;
    if (tid < 16) scale = input[(size_t)(rbase + tid) * ROW_STRIDE + T_STEPS * NINP];

    const v8f vzero = {0.f, 0.f, 0.f, 0.f, 0.f, 0.f, 0.f, 0.f};
    float cP[2][8] = {}, hP[2][8];
    __syncthreads();

    // ------------------------- Phase 1: pia LSTM -------------------------
    for (int t = 0; t < T_STEPS; ++t) {
        for (int i = tid; i < 16 * NINP; i += 512) {
            int r = i >> 6, c = i & 63;
            s_x[r * XPAD + c] =
                (bf16_t)input[(size_t)(rbase + r) * ROW_STRIDE + t * NINP + c];
        }
        __syncthreads();

        // Opaque zero offset per step: weight-fragment loads stay inside the
        // time loop (streamed from L2) but remain global_load (not flat).
        const bf16_t* wp = WpP + opaque_zero();

        v8f acc[8];
#pragma unroll
        for (int j = 0; j < 8; ++j) acc[j] = vzero;
        mm_acc(wp, 18, 0, 2,  s_x,  XPAD, lane, tiles, acc);   // x @ Wih_p.T
        mm_acc(wp, 18, 2, 16, s_hA, NH,   lane, tiles, acc);   // h @ Whh_p.T
        lstm_gates(acc, biasP, cP, hP);
        dot_partial(s_dot, lane, wl0, wl1, hP);
        __syncthreads();                       // GEMM reads + atomics done

        store_h(s_hA, lane, w, hP);
        if (tid < 16) {
            float v = s_dot[tid] + bl0;        // outs_p[t][row]
            s_pia[t * 16 + tid] = fabsf(v) * scale;
            if (t == T_STEPS - 1)
                out[(size_t)(rbase + tid) * (T_STEPS + 1) + T_STEPS] = v;  // last_pia
            s_dot[tid] = 0.0f;
        }
        __syncthreads();                       // new h visible
    }

    // ---------------------- Phase 2: 2-layer LSTM ------------------------
    for (int i = tid; i < 16 * NH; i += 512) {
        s_hA[i] = (bf16_t)0.0f;                // h1 = 0
        s_hB[i] = (bf16_t)0.0f;                // h2 = 0
    }
    for (int i = tid; i < 16 * 31; i += 512) { // zero pad cols 65..95 once
        int r = i / 31, c = 65 + i % 31;
        s_x[r * XPAD + c] = (bf16_t)0.0f;
    }
    float c1[2][8] = {}, c2[2][8] = {}, h1v[2][8], h2v[2][8];
    __syncthreads();

    for (int t = 0; t < T_STEPS; ++t) {
        for (int i = tid; i < 16 * NINP; i += 512) {
            int r = i >> 6, c = i & 63;
            s_x[r * XPAD + c] =
                (bf16_t)input[(size_t)(rbase + r) * ROW_STRIDE + t * NINP + c];
        }
        if (tid < 16)                          // x = [x_t, pia_t]
            s_x[tid * XPAD + NINP] = (bf16_t)s_pia[t * 16 + tid];
        __syncthreads();

        const bf16_t* wp1 = Wp1 + opaque_zero();
        const bf16_t* wp2 = Wp2 + opaque_zero();

        // layer 1: [x,pia] @ Wih1.T + h1 @ Whh1.T
        v8f acc[8];
#pragma unroll
        for (int j = 0; j < 8; ++j) acc[j] = vzero;
        mm_acc(wp1, 19, 0, 3,  s_x,  XPAD, lane, tiles, acc);
        mm_acc(wp1, 19, 3, 16, s_hA, NH,   lane, tiles, acc);
        lstm_gates(acc, bias1, c1, h1v);
        __syncthreads();                       // old-h1 reads done
        store_h(s_hA, lane, w, h1v);
        __syncthreads();                       // new h1 visible

        // layer 2: h1 @ Wih2.T + h2 @ Whh2.T
#pragma unroll
        for (int j = 0; j < 8; ++j) acc[j] = vzero;
        mm_acc(wp2, 32, 0,  16, s_hA, NH, lane, tiles, acc);
        mm_acc(wp2, 32, 16, 16, s_hB, NH, lane, tiles, acc);
        lstm_gates(acc, bias2, c2, h2v);
        dot_partial(s_dot, lane, wl0, wl1, h2v);
        __syncthreads();                       // old-h2 reads + atomics done

        store_h(s_hB, lane, w, h2v);
        if (tid < 16) {
            out[(size_t)(rbase + tid) * (T_STEPS + 1) + t] = s_dot[tid] + bl0;
            s_dot[tid] = 0.0f;
        }
        __syncthreads();                       // new h2 visible
    }
}

// ---------------------------------------------------------------------------
// launcher
// ---------------------------------------------------------------------------
extern "C" void kernel_launch(void* const* d_in, const int* in_sizes, int n_in,
                              void* d_out, int out_size, void* d_ws, size_t ws_size,
                              hipStream_t stream)
{
    const float* input = (const float*)d_in[0];
    const float* WihP  = (const float*)d_in[1];
    const float* WhhP  = (const float*)d_in[2];
    const float* bihP  = (const float*)d_in[3];
    const float* bhhP  = (const float*)d_in[4];
    const float* Wih1  = (const float*)d_in[5];
    const float* Whh1  = (const float*)d_in[6];
    const float* bih1  = (const float*)d_in[7];
    const float* bhh1  = (const float*)d_in[8];
    const float* Wih2  = (const float*)d_in[9];
    const float* Whh2  = (const float*)d_in[10];
    const float* bih2  = (const float*)d_in[11];
    const float* bhh2  = (const float*)d_in[12];
    const float* Wl    = (const float*)d_in[13];
    const float* bl    = (const float*)d_in[14];

    // workspace layout (bf16 packed weights + fp32 bias sums), ~9.1 MB total
    const size_t szP = (size_t)NTILES * 18 * 512;   // pia:    Kin 64  -> KT 18
    const size_t sz1 = (size_t)NTILES * 19 * 512;   // layer1: Kin 65  -> KT 19
    const size_t sz2 = (size_t)NTILES * 32 * 512;   // layer2: Kin 512 -> KT 32
    bf16_t* wpP = (bf16_t*)d_ws;
    bf16_t* wp1 = wpP + szP;
    bf16_t* wp2 = wp1 + sz1;
    float*  bsum = (float*)(wp2 + sz2);             // 3*2048 floats

    pack_weights_kernel<<<(int)((szP + 255) / 256), 256, 0, stream>>>(
        WihP, WhhP, 64, 2, 18, wpP);
    pack_weights_kernel<<<(int)((sz1 + 255) / 256), 256, 0, stream>>>(
        Wih1, Whh1, 65, 3, 19, wp1);
    pack_weights_kernel<<<(int)((sz2 + 255) / 256), 256, 0, stream>>>(
        Wih2, Whh2, 512, 16, 32, wp2);
    pack_bias_kernel<<<24, 256, 0, stream>>>(bihP, bhhP, bih1, bhh1, bih2, bhh2, bsum);

    lstm_stack_kernel<<<128, 512, 0, stream>>>(
        input, wpP, wp1, wp2, bsum, Wl, bl, (float*)d_out);
}
